// TransformerSpillNet_84207128805752
// MI455X (gfx1250) — matrix operand order repaired and checked by reading, not compiled, not run on hardware
//
#include <hip/hip_runtime.h>
#include <hip/hip_bf16.h>
#include <math.h>

// ---------------------------------------------------------------------------
// Problem constants (match reference)
// ---------------------------------------------------------------------------
#define NNODES  10000
#define NEDGES  160000
#define NGRAPHS 8
#define HID     256
#define HEADS   8
#define NL      4
#define FEAT    31
#define EDIM    32      // HID / HEADS

typedef __attribute__((ext_vector_type(16))) _Float16 v16h;
typedef __attribute__((ext_vector_type(8)))  _Float16 v8h;
typedef __attribute__((ext_vector_type(8)))  float    v8f;

// ---------------------------------------------------------------------------
// Small device helpers
// ---------------------------------------------------------------------------
__device__ __forceinline__ float gelu_exact(float x) {
    return 0.5f * x * (1.0f + erff(x * 0.70710678118654752f));
}

__device__ __forceinline__ float warp_sum32(float v) {
    #pragma unroll
    for (int o = 16; o > 0; o >>= 1) v += __shfl_xor(v, o, 32);
    return v;
}

// block reduce (blockDim.x multiple of 32, <=256); sm must hold >=8 floats
__device__ __forceinline__ float block_sum(float v, float* sm) {
    int t = (int)threadIdx.x;
    float w = warp_sum32(v);
    if ((t & 31) == 0) sm[t >> 5] = w;
    __syncthreads();
    float r = (t < (int)(blockDim.x >> 5)) ? sm[t] : 0.0f;
    if (t < 32) {
        r = warp_sum32(r);
        if (t == 0) sm[0] = r;
    }
    __syncthreads();
    float out = sm[0];
    __syncthreads();
    return out;
}

// order-preserving f32 <-> u32 mapping for atomicMax-based segment_max
__device__ __forceinline__ unsigned enc_f32(float f) {
    unsigned u = __float_as_uint(f);
    return (u & 0x80000000u) ? ~u : (u | 0x80000000u);
}
__device__ __forceinline__ float dec_f32(unsigned u) {
    return __uint_as_float((u & 0x80000000u) ? (u & 0x7FFFFFFFu) : ~u);
}

// ---------------------------------------------------------------------------
// Precision-prep kernels: f32 -> f16, and weight transpose+convert
// ---------------------------------------------------------------------------
__global__ void f32_to_f16(const float* __restrict__ in,
                           _Float16* __restrict__ out, int n) {
    int i = (int)blockIdx.x * (int)blockDim.x + (int)threadIdx.x;
    if (i < n) out[i] = (_Float16)in[i];
}

// W[K,N] f32 (row-major) -> Wt[N,K] f16 (row-major)
__global__ void transpose_f16(const float* __restrict__ W,
                              _Float16* __restrict__ Wt, int K, int N) {
    int i = (int)blockIdx.x * (int)blockDim.x + (int)threadIdx.x;
    if (i >= K * N) return;
    int k = i / N, n = i - k * N;
    Wt[(size_t)n * K + k] = (_Float16)W[i];
}

// ---------------------------------------------------------------------------
// WMMA GEMM (CDNA5 wave32):
//   C[M,N] = A[M,K] (f16, row-major) x Wt[N,K]^T (f16, pre-transposed) + bias
// Block = 256 threads (8 waves). Block tile = 16(M) x 128(N); each wave owns
// one 16x16 N-subtile. The shared 16x32 A chunk is staged into LDS with
// GLOBAL_LOAD_ASYNC_TO_LDS_B128 (ASYNCcnt / s_wait_asynccnt), then each lane
// builds its fragments with two ds_load_b128 (A) and two global_load_b128 (B).
// Fragment layouts per CDNA5 ISA 7.12.2. Requires M%16==0, N%128==0, K%32==0.
// ---------------------------------------------------------------------------
template <typename OutT>
__global__ void wmma_gemm_f16(const _Float16* __restrict__ A,
                              const _Float16* __restrict__ Wt,
                              const float* __restrict__ bias,
                              OutT* __restrict__ C,
                              int M, int N, int K) {
    __shared__ _Float16 smA[16 * 32];          // 1 KB staging tile
    const int t    = (int)threadIdx.x;
    const int lane = t & 31;
    const int wv   = t >> 5;                   // wave id 0..7
    const int m0   = (int)blockIdx.x * 16;
    const int n0   = (int)blockIdx.y * 128 + wv * 16;
    const int row  = lane & 15;                // A row within tile
    const int grp  = lane >> 4;                // lane group -> K sub-range
    const int n    = n0 + (lane & 15);

    v8f acc = {};
    for (int kt = 0; kt < K; kt += 32) {
        // ---- async DMA: stage A chunk (16 rows x 32 halves) to LDS ----
        if (t < 64) {                           // waves 0,1: 64 lanes x 16B = 1KB
            const int r   = t >> 2;             // 0..15
            const int c8  = (t & 3) * 8;        // half-offset 0,8,16,24
            unsigned lds = (unsigned)(size_t)smA + (unsigned)(r * 64 + c8 * 2);
            unsigned long long ga =
                (unsigned long long)(size_t)(A + (size_t)(m0 + r) * K + kt + c8);
            asm volatile("global_load_async_to_lds_b128 %0, %1, off"
                         :: "v"(lds), "v"(ga) : "memory");
            asm volatile("s_wait_asynccnt 0x0" ::: "memory");
        }
        __syncthreads();

        // ---- A fragment from LDS (ds_load_b128 x2) ----
        // grp0: halves 0..7 -> K kt+0..7,  8..15 -> K kt+16..23
        // grp1: halves 0..7 -> K kt+8..15, 8..15 -> K kt+24..31
        const _Float16* ar = smA + row * 32 + grp * 8;
        v8h alo = *(const v8h*)(ar);
        v8h ahi = *(const v8h*)(ar + 16);
        v16h a;
        #pragma unroll
        for (int j = 0; j < 8; ++j) { a[j] = alo[j]; a[8 + j] = ahi[j]; }

        // ---- B fragment from transposed weights (global_load_b128 x2) ----
        // halves 0..7 -> K kt+8*grp+0..7, 8..15 -> K kt+16+8*grp+0..7 (matches A)
        const _Float16* bp = Wt + (size_t)n * K + kt + grp * 8;
        v8h blo = *(const v8h*)(bp);
        v8h bhi = *(const v8h*)(bp + 16);
        v16h b;
        #pragma unroll
        for (int j = 0; j < 8; ++j) { b[j] = blo[j]; b[8 + j] = bhi[j]; }

        acc = __builtin_amdgcn_wmma_f32_16x16x32_f16(
            false, a, false, b, (short)0, acc, false, false);
        __syncthreads();                        // protect smA before next stage
    }

    const float bv = bias ? bias[n] : 0.0f;
    #pragma unroll
    for (int r = 0; r < 8; ++r) {               // C/D: grp0 rows 0..7, grp1 8..15
        C[(size_t)(m0 + grp * 8 + r) * N + n] = (OutT)(acc[r] + bv);
    }
}

// ---------------------------------------------------------------------------
// Input encoder: h = GELU(LN(x @ W + b)); one block (256 thr) per node
// ---------------------------------------------------------------------------
__global__ void input_encoder(const float* __restrict__ x,
                              const float* __restrict__ W,
                              const float* __restrict__ b,
                              const float* __restrict__ g,
                              const float* __restrict__ be,
                              float* __restrict__ h) {
    __shared__ float xr[FEAT];
    __shared__ float red[8];
    const int n = (int)blockIdx.x, t = (int)threadIdx.x;
    if (t < FEAT) xr[t] = x[(size_t)n * FEAT + t];
    __syncthreads();
    float acc = b[t];
    #pragma unroll
    for (int f = 0; f < FEAT; ++f) acc += xr[f] * W[f * HID + t];
    const float mean = block_sum(acc, red) * (1.0f / HID);
    const float d    = acc - mean;
    const float var  = block_sum(d * d, red) * (1.0f / HID);
    const float y    = d * rsqrtf(var + 1e-5f) * g[t] + be[t];
    h[(size_t)n * HID + t] = gelu_exact(y);
}

// ---------------------------------------------------------------------------
// Edge encoder: e = GELU(LN(edge_attr @ W + b)); one wave per edge (32 ch)
// ---------------------------------------------------------------------------
__global__ void edge_encoder(const float* __restrict__ ea,
                             const float* __restrict__ W,
                             const float* __restrict__ b,
                             const float* __restrict__ g,
                             const float* __restrict__ be,
                             float* __restrict__ e) {
    const int wid = (int)threadIdx.x >> 5, lane = (int)threadIdx.x & 31;
    const int idx = (int)blockIdx.x * 8 + wid;
    if (idx >= NEDGES) return;
    const float a   = ea[idx];
    const float pre = a * W[lane] + b[lane];
    const float mean = warp_sum32(pre) * (1.0f / EDIM);
    const float d    = pre - mean;
    const float var  = warp_sum32(d * d) * (1.0f / EDIM);
    const float y    = d * rsqrtf(var + 1e-5f) * g[lane] + be[lane];
    e[(size_t)idx * EDIM + lane] = gelu_exact(y);
}

// ---------------------------------------------------------------------------
// Attention scores + segment max (one wave per edge)
// ---------------------------------------------------------------------------
__global__ void score_kernel(const float* __restrict__ q,
                             const float* __restrict__ k,
                             const _Float16* __restrict__ ee,
                             const int* __restrict__ src,
                             const int* __restrict__ dst,
                             float* __restrict__ score,
                             unsigned* __restrict__ menc,
                             int H) {
    const int wid = (int)threadIdx.x >> 5, lane = (int)threadIdx.x & 31;
    const int e = (int)blockIdx.x * 8 + wid;
    if (e >= NEDGES) return;
    const int s = src[e], d = dst[e];
    const int C = HID / H;
    const float scale = rsqrtf((float)C);
    for (int hd = 0; hd < H; ++hd) {
        float part = 0.0f;
        for (int c = lane; c < C; c += 32) {
            const int ch = hd * C + c;
            const float kj = k[(size_t)s * HID + ch] + (float)ee[(size_t)e * HID + ch];
            part += q[(size_t)d * HID + ch] * kj;
        }
        const float sc = warp_sum32(part) * scale;
        if (lane == 0) {
            score[(size_t)e * H + hd] = sc;
            atomicMax(&menc[(size_t)d * H + hd], enc_f32(sc));
        }
    }
}

// exp(score - m[dst]) and segment-sum denominator
__global__ void softmax_num_den(const int* __restrict__ dst,
                                float* __restrict__ score,
                                const unsigned* __restrict__ menc,
                                float* __restrict__ den, int H) {
    const int i = (int)blockIdx.x * (int)blockDim.x + (int)threadIdx.x;
    if (i >= NEDGES * H) return;
    const int e = i / H, hd = i - e * H;
    const int d = dst[e];
    const float m  = dec_f32(menc[(size_t)d * H + hd]);
    const float ex = expf(score[i] - m);
    score[i] = ex;
    atomicAdd(&den[(size_t)d * H + hd], ex);
}

// scatter-aggregate alpha * (v[src] + ee) into destination nodes
__global__ void aggregate_kernel(const float* __restrict__ v,
                                 const _Float16* __restrict__ ee,
                                 const int* __restrict__ src,
                                 const int* __restrict__ dst,
                                 const float* __restrict__ score,
                                 const float* __restrict__ den,
                                 float* __restrict__ agg, int H) {
    const int wid = (int)threadIdx.x >> 5, lane = (int)threadIdx.x & 31;
    const int e = (int)blockIdx.x * 8 + wid;
    if (e >= NEDGES) return;
    const int s = src[e], d = dst[e];
    const int C = HID / H;
    for (int c = lane; c < HID; c += 32) {
        const int hd = c / C;
        const float alpha = score[(size_t)e * H + hd] /
                            fmaxf(den[(size_t)d * H + hd], 1e-16f);
        const float vj = v[(size_t)s * HID + c] + (float)ee[(size_t)e * HID + c];
        atomicAdd(&agg[(size_t)d * HID + c], alpha * vj);
    }
}

// beta gate + LN + GELU + 0.5*residual; one block (256 thr) per node
__global__ void gate_ln_res(const float* __restrict__ agg,
                            const float* __restrict__ root,
                            const float* __restrict__ Wb,
                            const float* __restrict__ ng,
                            const float* __restrict__ nb,
                            const float* __restrict__ hres,
                            float* __restrict__ hout) {
    __shared__ float red[8];
    const int n = (int)blockIdx.x, t = (int)threadIdx.x;
    const float o = agg[(size_t)n * HID + t];
    const float r = root[(size_t)n * HID + t];
    const float part = o * Wb[t] + r * Wb[HID + t] + (o - r) * Wb[2 * HID + t];
    const float gl   = block_sum(part, red);
    const float gsig = 1.0f / (1.0f + expf(-gl));
    const float hv   = gsig * r + (1.0f - gsig) * o;
    const float mean = block_sum(hv, red) * (1.0f / HID);
    const float dd   = hv - mean;
    const float var  = block_sum(dd * dd, red) * (1.0f / HID);
    const float y    = dd * rsqrtf(var + 1e-5f) * ng[t] + nb[t];
    hout[(size_t)n * HID + t] = gelu_exact(y) + 0.5f * hres[(size_t)n * HID + t];
}

// ---------------------------------------------------------------------------
// Pooling
// ---------------------------------------------------------------------------
__global__ void pool_kernel(const float* __restrict__ h,
                            const int* __restrict__ batch,
                            float* __restrict__ ssum,
                            unsigned* __restrict__ mxenc,
                            float* __restrict__ cnt) {
    const int i = (int)blockIdx.x * (int)blockDim.x + (int)threadIdx.x;
    if (i >= NNODES * HID) return;
    const int n = i / HID, c = i - n * HID;
    const int g = batch[n];
    const float v = h[i];
    atomicAdd(&ssum[g * HID + c], v);
    atomicMax(&mxenc[g * HID + c], enc_f32(v));
    if (c == 0) atomicAdd(&cnt[g], 1.0f);
}

__global__ void make_pooled(const float* __restrict__ ssum,
                            const unsigned* __restrict__ mxenc,
                            const float* __restrict__ cnt,
                            float* __restrict__ pooled) {
    const int i = (int)blockIdx.x * (int)blockDim.x + (int)threadIdx.x;
    if (i >= NGRAPHS * 3 * HID) return;
    const int g = i / (3 * HID), c = i - g * 3 * HID;
    float v;
    if (c < HID)            v = ssum[g * HID + c] / cnt[g];
    else if (c < 2 * HID)   v = dec_f32(mxenc[g * HID + c - HID]);
    else                    v = ssum[g * HID + c - 2 * HID];
    pooled[i] = v;
}

// ---------------------------------------------------------------------------
// Prediction head (8 rows -> scalar work)
// ---------------------------------------------------------------------------
__global__ void head1(const float* __restrict__ pooled, const float* __restrict__ W1,
                      const float* __restrict__ b1, const float* __restrict__ g1,
                      const float* __restrict__ be1, float* __restrict__ t1) {
    __shared__ float pr[3 * HID];
    __shared__ float red[8];
    const int g = (int)blockIdx.x, t = (int)threadIdx.x;
    for (int j = t; j < 3 * HID; j += 256) pr[j] = pooled[g * 3 * HID + j];
    __syncthreads();
    float acc = b1[t];
    for (int kk = 0; kk < 3 * HID; ++kk) acc += pr[kk] * W1[kk * HID + t];
    const float mean = block_sum(acc, red) * (1.0f / HID);
    const float d    = acc - mean;
    const float var  = block_sum(d * d, red) * (1.0f / HID);
    const float y    = d * rsqrtf(var + 1e-5f) * g1[t] + be1[t];
    t1[g * HID + t] = gelu_exact(y);
}

__global__ void head2(const float* __restrict__ t1, const float* __restrict__ W2,
                      const float* __restrict__ b2, float* __restrict__ t2) {
    __shared__ float pr[HID];
    const int g = (int)blockIdx.x, t = (int)threadIdx.x;  // 128 threads
    for (int j = t; j < HID; j += 128) pr[j] = t1[g * HID + j];
    __syncthreads();
    float acc = b2[t];
    for (int kk = 0; kk < HID; ++kk) acc += pr[kk] * W2[kk * 128 + t];
    t2[g * 128 + t] = gelu_exact(acc);
}

__global__ void head3(const float* __restrict__ t2, const float* __restrict__ W3,
                      const float* __restrict__ b3, const float* __restrict__ g3,
                      const float* __restrict__ be3, float* __restrict__ t3) {
    __shared__ float pr[128];
    __shared__ float red[8];
    const int g = (int)blockIdx.x, t = (int)threadIdx.x;  // 64 threads
    for (int j = t; j < 128; j += 64) pr[j] = t2[g * 128 + j];
    __syncthreads();
    float acc = b3[t];
    for (int kk = 0; kk < 128; ++kk) acc += pr[kk] * W3[kk * 64 + t];
    const float mean = block_sum(acc, red) * (1.0f / 64.0f);
    const float d    = acc - mean;
    const float var  = block_sum(d * d, red) * (1.0f / 64.0f);
    const float y    = d * rsqrtf(var + 1e-5f) * g3[t] + be3[t];
    t3[g * 64 + t] = gelu_exact(y);
}

__global__ void head4(const float* __restrict__ t3, const float* __restrict__ W4,
                      const float* __restrict__ b4, float* __restrict__ gout) {
    const int g = (int)threadIdx.x;
    if (g >= NGRAPHS) return;
    float acc = b4[0];
    for (int kk = 0; kk < 64; ++kk) acc += t3[g * 64 + kk] * W4[kk];
    gout[g] = acc;
}

__global__ void broadcast_out(const float* __restrict__ gout,
                              const int* __restrict__ batch,
                              float* __restrict__ out) {
    const int n = (int)blockIdx.x * (int)blockDim.x + (int)threadIdx.x;
    if (n >= NNODES) return;
    out[n] = gout[batch[n]];
}

// fills
__global__ void fill_f32(float* p, float v, int n) {
    int i = (int)blockIdx.x * (int)blockDim.x + (int)threadIdx.x;
    if (i < n) p[i] = v;
}
__global__ void fill_u32(unsigned* p, unsigned v, int n) {
    int i = (int)blockIdx.x * (int)blockDim.x + (int)threadIdx.x;
    if (i < n) p[i] = v;
}

// ---------------------------------------------------------------------------
// Host orchestration
// ---------------------------------------------------------------------------
extern "C" void kernel_launch(void* const* d_in, const int* in_sizes, int n_in,
                              void* d_out, int out_size, void* d_ws, size_t ws_size,
                              hipStream_t stream) {
    (void)in_sizes; (void)n_in; (void)out_size; (void)ws_size;

    // ---- inputs (setup_inputs dict order, nested dicts in insertion order) ----
    const float* x     = (const float*)d_in[0];
    const int*   eidx  = (const int*)  d_in[1];
    const float* eattr = (const float*)d_in[2];
    const int*   batch = (const int*)  d_in[3];
    const int*   src   = eidx;              // edge_index[0]
    const int*   dst   = eidx + NEDGES;     // edge_index[1]
    auto F = [&](int i) { return (const float*)d_in[i]; };
    // 4..7:  inp W,b,g,be   8..11: edge W,b,g,be
    // 12+12l+{0..11}: Wq,bq,Wk,bk,Wv,bv,We,Ws,bs,Wb,ng,nb
    // 60..65: pred W1,b1,g1,be1,W2,b2    66..71: out W3,b3,g3,be3,W4,b4

    // ---- workspace carve-up ----
    char* base = (char*)d_ws;
    size_t off = 0;
    auto alloc = [&](size_t bytes) -> void* {
        off = (off + 255) & ~(size_t)255;
        void* p = base + off;
        off += bytes;
        return p;
    };
    float*     h1    = (float*)alloc((size_t)NNODES * HID * 4);
    float*     h2    = (float*)alloc((size_t)NNODES * HID * 4);
    float*     ebuf  = (float*)alloc((size_t)NEDGES * EDIM * 4);
    _Float16*  hf16  = (_Float16*)alloc((size_t)NNODES * HID * 2);
    _Float16*  ef16  = (_Float16*)alloc((size_t)NEDGES * EDIM * 2);
    float*     q     = (float*)alloc((size_t)NNODES * HID * 4);
    float*     k     = (float*)alloc((size_t)NNODES * HID * 4);
    float*     v     = (float*)alloc((size_t)NNODES * HID * 4);
    float*     root  = (float*)alloc((size_t)NNODES * HID * 4);
    _Float16*  ee    = (_Float16*)alloc((size_t)NEDGES * HID * 2);
    float*     score = (float*)alloc((size_t)NEDGES * HEADS * 4);
    unsigned*  menc  = (unsigned*)alloc((size_t)NNODES * HEADS * 4);
    float*     den   = (float*)alloc((size_t)NNODES * HEADS * 4);
    float*     agg   = (float*)alloc((size_t)NNODES * HID * 4);
    float*     ssum  = (float*)alloc((size_t)NGRAPHS * HID * 4);
    unsigned*  mxenc = (unsigned*)alloc((size_t)NGRAPHS * HID * 4);
    float*     cnt   = (float*)alloc((size_t)NGRAPHS * 4);
    float*     pooled= (float*)alloc((size_t)NGRAPHS * 3 * HID * 4);
    float*     t1    = (float*)alloc((size_t)NGRAPHS * HID * 4);
    float*     t2    = (float*)alloc((size_t)NGRAPHS * 128 * 4);
    float*     t3    = (float*)alloc((size_t)NGRAPHS * 64 * 4);
    float*     gout  = (float*)alloc((size_t)NGRAPHS * 4);
    // transposed f16 weight arena: per layer 4x[256,256] + 1x[256,32]
    _Float16*  WtQ[NL]; _Float16* WtK[NL]; _Float16* WtV[NL];
    _Float16*  WtS[NL]; _Float16* WtE[NL];
    for (int l = 0; l < NL; ++l) {
        WtQ[l] = (_Float16*)alloc((size_t)HID * HID * 2);
        WtK[l] = (_Float16*)alloc((size_t)HID * HID * 2);
        WtV[l] = (_Float16*)alloc((size_t)HID * HID * 2);
        WtS[l] = (_Float16*)alloc((size_t)HID * HID * 2);
        WtE[l] = (_Float16*)alloc((size_t)HID * EDIM * 2);
    }

    auto gemm_f32out = [&](const _Float16* A, const _Float16* Wt, const float* bias,
                           float* C, int M, int N, int K) {
        dim3 g(M / 16, N / 128);
        wmma_gemm_f16<float><<<g, 256, 0, stream>>>(A, Wt, bias, C, M, N, K);
    };
    auto gemm_f16out = [&](const _Float16* A, const _Float16* Wt, _Float16* C,
                           int M, int N, int K) {
        dim3 g(M / 16, N / 128);
        wmma_gemm_f16<_Float16><<<g, 256, 0, stream>>>(A, Wt, nullptr, C, M, N, K);
    };

    // ---- one-time weight transpose+convert (f32 [K,N] -> f16 [N,K]) ----
    for (int l = 0; l < NL; ++l) {
        const int p = 12 + 12 * l;
        const int nHH = HID * HID, nEH = EDIM * HID;
        transpose_f16<<<(nHH + 255) / 256, 256, 0, stream>>>(F(p + 0), WtQ[l], HID, HID);
        transpose_f16<<<(nHH + 255) / 256, 256, 0, stream>>>(F(p + 2), WtK[l], HID, HID);
        transpose_f16<<<(nHH + 255) / 256, 256, 0, stream>>>(F(p + 4), WtV[l], HID, HID);
        transpose_f16<<<(nHH + 255) / 256, 256, 0, stream>>>(F(p + 7), WtS[l], HID, HID);
        transpose_f16<<<(nEH + 255) / 256, 256, 0, stream>>>(F(p + 6), WtE[l], EDIM, HID);
    }

    // ---- encoders ----
    input_encoder<<<NNODES, 256, 0, stream>>>(x, F(4), F(5), F(6), F(7), h1);
    edge_encoder<<<NEDGES / 8, 256, 0, stream>>>(eattr, F(8), F(9), F(10), F(11), ebuf);
    f32_to_f16<<<(NEDGES * EDIM + 255) / 256, 256, 0, stream>>>(ebuf, ef16, NEDGES * EDIM);

    float* hcur = h1;
    float* hnext = h2;

    // ---- transformer layers ----
    for (int l = 0; l < NL; ++l) {
        const int p = 12 + 12 * l;
        const int H = (l < NL - 1) ? HEADS : 1;

        f32_to_f16<<<(NNODES * HID + 255) / 256, 256, 0, stream>>>(hcur, hf16, NNODES * HID);

        gemm_f32out(hf16, WtQ[l], F(p + 1), q,    NNODES, HID, HID);   // Q
        gemm_f32out(hf16, WtK[l], F(p + 3), k,    NNODES, HID, HID);   // K
        gemm_f32out(hf16, WtV[l], F(p + 5), v,    NNODES, HID, HID);   // V
        gemm_f32out(hf16, WtS[l], F(p + 8), root, NNODES, HID, HID);   // skip/root
        gemm_f16out(ef16, WtE[l], ee, NEDGES, HID, EDIM);              // edge emb

        fill_u32<<<(NNODES * HEADS + 255) / 256, 256, 0, stream>>>(menc, 0u, NNODES * HEADS);
        fill_f32<<<(NNODES * HEADS + 255) / 256, 256, 0, stream>>>(den, 0.0f, NNODES * HEADS);
        fill_f32<<<(NNODES * HID + 255) / 256, 256, 0, stream>>>(agg, 0.0f, NNODES * HID);

        score_kernel<<<NEDGES / 8, 256, 0, stream>>>(q, k, ee, src, dst, score, menc, H);
        softmax_num_den<<<(NEDGES * H + 255) / 256, 256, 0, stream>>>(dst, score, menc, den, H);
        aggregate_kernel<<<NEDGES / 8, 256, 0, stream>>>(v, ee, src, dst, score, den, agg, H);

        gate_ln_res<<<NNODES, 256, 0, stream>>>(agg, root, F(p + 9), F(p + 10), F(p + 11),
                                                hcur, hnext);
        float* tmp = hcur; hcur = hnext; hnext = tmp;
    }

    // ---- pooling ----
    fill_f32<<<(NGRAPHS * HID + 255) / 256, 256, 0, stream>>>(ssum, 0.0f, NGRAPHS * HID);
    fill_u32<<<(NGRAPHS * HID + 255) / 256, 256, 0, stream>>>(mxenc, 0u, NGRAPHS * HID);
    fill_f32<<<1, 32, 0, stream>>>(cnt, 0.0f, NGRAPHS);
    pool_kernel<<<(NNODES * HID + 255) / 256, 256, 0, stream>>>(hcur, batch, ssum, mxenc, cnt);
    make_pooled<<<(NGRAPHS * 3 * HID + 255) / 256, 256, 0, stream>>>(ssum, mxenc, cnt, pooled);

    // ---- prediction head ----
    head1<<<NGRAPHS, 256, 0, stream>>>(pooled, F(60), F(61), F(62), F(63), t1);
    head2<<<NGRAPHS, 128, 0, stream>>>(t1, F(64), F(65), t2);
    head3<<<NGRAPHS, 64, 0, stream>>>(t2, F(66), F(67), F(68), F(69), t3);
    head4<<<1, 32, 0, stream>>>(t3, F(70), F(71), gout);
    broadcast_out<<<(NNODES + 255) / 256, 256, 0, stream>>>(gout, batch, (float*)d_out);
}
